// Model_77455440216368
// MI455X (gfx1250) — matrix operand order repaired and checked by reading, not compiled, not run on hardware
//
#include <hip/hip_runtime.h>
#include <hip/hip_bf16.h>

// ---------------- types for WMMA ----------------
typedef __attribute__((ext_vector_type(16))) _Float16 v16h;
typedef __attribute__((ext_vector_type(8)))  _Float16 v8h;
typedef __attribute__((ext_vector_type(8)))  float    v8f;
typedef __attribute__((ext_vector_type(4)))  float    v4f;

union FragAB { v16h v; v8h h[2]; };

// ---------------- model constants ----------------
#define QLEN   512
#define BSZ    2
#define MLEN   1024
#define KLEN   1536
#define DMODEL 768
#define NHEAD  12
#define DHEAD  64
#define DINNER 3072
#define NLAYER 12
#define WORDS  249
#define EPSF   1e-5f

// ---------------- workspace layout (float offsets) ----------------
#define O_CAT   0LL                      // (1536*2, 768)
#define O_HEADS (O_CAT   + 2359296LL)    // (3072, 2304)
#define O_QAC   (O_HEADS + 7077888LL)    // (2,12,512,64)
#define O_QBD   (O_QAC   + 786432LL)
#define O_KB    (O_QBD   + 786432LL)     // (2,12,1536,64)
#define O_VB    (O_KB    + 2359296LL)    // (2,12,64,1536)  V stored transposed
#define O_RK    (O_VB    + 2359296LL)    // (1536, 768)
#define O_AC    (O_RK    + 1179648LL)    // (2,12,512,1536)  (prob reuses this)
#define O_BD    (O_AC    + 18874368LL)
#define O_VEC   (O_BD    + 18874368LL)   // (1024, 768) laid out (i,bi,n,d)
#define O_H     (O_VEC   + 786432LL)     // (1024, 768)
#define O_TMP   (O_H     + 786432LL)     // attnout / ff2out / final LN
#define O_FF    (O_TMP   + 786432LL)     // (1024, 3072); also reused as embed gather E
#define O_PE    (O_FF    + 3145728LL)    // (1536, 768)
// total ~= 61.3M floats (~245 MB)

// ---------------- generic batched WMMA GEMM ----------------
// C = alpha * A(MxK) * B(N,K)^T + bias[N] + res   (B row-major (N,K), K contiguous)
// Block tile 128x128, 8 waves of 32x64 -> 8x v_wmma_f32_16x16x32_f16 per K-slab.
// Interior tiles: branch-free float4 staging; edge tiles: clamped loads + select.
#define BM 128
#define BN 128
#define BK 32
#define LDSROW 40   // halves per LDS row (32 + 8 pad) = 80B, keeps 16B chunk alignment

// stage one 128x32 f32 tile (rows r0..r0+127, cols k0..k0+31 of a rowsxK matrix,
// leading dim ld) into LDS as f16 [row][k]; each thread: 16 floats.
__device__ __forceinline__ void stage_tile(const float* __restrict__ S, int rows, int K,
                                           int ld, int r0, int k0, bool edge,
                                           _Float16* __restrict__ lds, int r, int c0) {
    if (!edge) {
        const float* src = S + (long long)(r0 + r) * ld + k0 + c0;
        const v4f f0 = *(const v4f*)src;
        const v4f f1 = *(const v4f*)(src + 4);
        const v4f f2 = *(const v4f*)(src + 8);
        const v4f f3 = *(const v4f*)(src + 12);
        v8h t0, t1;
        t0[0] = (_Float16)f0[0]; t0[1] = (_Float16)f0[1];
        t0[2] = (_Float16)f0[2]; t0[3] = (_Float16)f0[3];
        t0[4] = (_Float16)f1[0]; t0[5] = (_Float16)f1[1];
        t0[6] = (_Float16)f1[2]; t0[7] = (_Float16)f1[3];
        t1[0] = (_Float16)f2[0]; t1[1] = (_Float16)f2[1];
        t1[2] = (_Float16)f2[2]; t1[3] = (_Float16)f2[3];
        t1[4] = (_Float16)f3[0]; t1[5] = (_Float16)f3[1];
        t1[6] = (_Float16)f3[2]; t1[7] = (_Float16)f3[3];
        *(v8h*)&lds[r * LDSROW + c0]     = t0;
        *(v8h*)&lds[r * LDSROW + c0 + 8] = t1;
    } else {
        const int  gr  = (r0 + r < rows) ? (r0 + r) : (rows - 1);
        const bool rok = (r0 + r) < rows;
        const float* src = S + (long long)gr * ld;
        v8h t0, t1;
        #pragma unroll
        for (int u = 0; u < 8; ++u) {
            const int gk = k0 + c0 + u;
            const int ck = (gk < K) ? gk : (K - 1);
            const float v = src[ck];
            t0[u] = (_Float16)((rok && gk < K) ? v : 0.0f);
        }
        #pragma unroll
        for (int u = 0; u < 8; ++u) {
            const int gk = k0 + c0 + 8 + u;
            const int ck = (gk < K) ? gk : (K - 1);
            const float v = src[ck];
            t1[u] = (_Float16)((rok && gk < K) ? v : 0.0f);
        }
        *(v8h*)&lds[r * LDSROW + c0]     = t0;
        *(v8h*)&lds[r * LDSROW + c0 + 8] = t1;
    }
}

__global__ __launch_bounds__(256)
void k_gemm(const float* __restrict__ A, const float* __restrict__ B,
            const float* __restrict__ bias, const float* __restrict__ res,
            float* __restrict__ C,
            int M, int N, int K, int lda, int ldb, int ldc,
            long long sA1, long long sB1, long long sC1, int nb1,
            long long sA2, long long sB2, long long sC2,
            int doRelu, float alpha)
{
    __shared__ __align__(16) _Float16 As[BM * LDSROW];
    __shared__ __align__(16) _Float16 Bs[BN * LDSROW];

    const int tid = threadIdx.x;
    const int z  = blockIdx.z;
    const int b1 = z % nb1, b2 = z / nb1;
    A += b1 * sA1 + b2 * sA2;
    B += b1 * sB1 + b2 * sB2;
    const long long co = b1 * sC1 + b2 * sC2;
    C += co;
    if (res) res += co;

    const int m0 = blockIdx.y * BM;
    const int n0 = blockIdx.x * BN;
    const bool edgeM = (m0 + BM > M);
    const bool edgeN = (n0 + BN > N);

    v8f acc[2][4] = {};

    const int wid = tid >> 5, lane = tid & 31;
    const int wm = (wid >> 1) * 32;        // 4 wave-rows of 32
    const int wn = (wid & 1) * 64;         // 2 wave-cols of 64
    const int lr  = lane & 15;
    const int akb = (lane < 16) ? 0 : 8;   // ISA 16-bit A 16x32 layout
    const int bkb = (lane < 16) ? 0 : 8;   // B 32x16 layout (same K order as A)

    const int sr = tid >> 1, sc0 = (tid & 1) * 16;  // staging: 128 rows x 32 cols

    for (int k0 = 0; k0 < K; k0 += BK) {
        const bool edgeK = (k0 + BK > K);
        __syncthreads();
        stage_tile(A, M, K, lda, m0, k0, edgeM || edgeK, As, sr, sc0);
        stage_tile(B, N, K, ldb, n0, k0, edgeN || edgeK, Bs, sr, sc0);
        __syncthreads();

        if (k0 + BK < K) { // hint next K-slab into cache (global_prefetch_b8)
            __builtin_prefetch(&A[(long long)(m0 + sr) * lda + k0 + BK], 0, 1);
            __builtin_prefetch(&B[(long long)(n0 + sr) * ldb + k0 + BK], 0, 1);
        }

        FragAB a[2], b[4];
        #pragma unroll
        for (int fm = 0; fm < 2; ++fm) {
            a[fm].h[0] = *(const v8h*)&As[(wm + fm * 16 + lr) * LDSROW + akb];
            a[fm].h[1] = *(const v8h*)&As[(wm + fm * 16 + lr) * LDSROW + akb + 16];
        }
        #pragma unroll
        for (int fn = 0; fn < 4; ++fn) {
            b[fn].h[0] = *(const v8h*)&Bs[(wn + fn * 16 + lr) * LDSROW + bkb];
            b[fn].h[1] = *(const v8h*)&Bs[(wn + fn * 16 + lr) * LDSROW + bkb + 16];
        }
        #pragma unroll
        for (int fm = 0; fm < 2; ++fm) {
            #pragma unroll
            for (int fn = 0; fn < 4; ++fn) {
                acc[fm][fn] = __builtin_amdgcn_wmma_f32_16x16x32_f16(
                    false, a[fm].v, false, b[fn].v, (short)0, acc[fm][fn], false, false);
            }
        }
    }

    // epilogue: C/D layout — VGPR r: M = r (+8 for lanes 16..31), N = lane%16
    const int rowadd = (lane < 16) ? 0 : 8;
    if (!edgeM && !edgeN) {
        #pragma unroll
        for (int fm = 0; fm < 2; ++fm) {
            #pragma unroll
            for (int fn = 0; fn < 4; ++fn) {
                const int col = n0 + wn + fn * 16 + lr;
                const float bv = bias ? bias[col] : 0.0f;
                #pragma unroll
                for (int r = 0; r < 8; ++r) {
                    const int row = m0 + wm + fm * 16 + r + rowadd;
                    float v = alpha * acc[fm][fn][r] + bv;
                    if (res) v += res[(long long)row * ldc + col];
                    if (doRelu) v = fmaxf(v, 0.0f);
                    C[(long long)row * ldc + col] = v;
                }
            }
        }
    } else {
        #pragma unroll
        for (int fm = 0; fm < 2; ++fm) {
            #pragma unroll
            for (int fn = 0; fn < 4; ++fn) {
                const int col = n0 + wn + fn * 16 + lr;
                if (col >= N) continue;
                const float bv = bias ? bias[col] : 0.0f;
                #pragma unroll
                for (int r = 0; r < 8; ++r) {
                    const int row = m0 + wm + fm * 16 + r + rowadd;
                    if (row < M) {
                        float v = alpha * acc[fm][fn][r] + bv;
                        if (res) v += res[(long long)row * ldc + col];
                        if (doRelu) v = fmaxf(v, 0.0f);
                        C[(long long)row * ldc + col] = v;
                    }
                }
            }
        }
    }
}

// ---------------- elementwise kernels ----------------
__global__ void k_zero(float* p, long long n) {
    long long i = (long long)blockIdx.x * blockDim.x + threadIdx.x;
    if (i < n) p[i] = 0.0f;
}
__global__ void k_copy(float* dst, const float* src, long long n) {
    long long i = (long long)blockIdx.x * blockDim.x + threadIdx.x;
    if (i < n) dst[i] = src[i];
}

// adaptive embedding gather: all tokens < 249 < cutoff0=20000 -> bucket 0 only.
__global__ void k_gather_embed(const int* __restrict__ x, const float* __restrict__ emb0,
                               float* __restrict__ E) {
    long long i = (long long)blockIdx.x * blockDim.x + threadIdx.x;
    if (i >= (long long)QLEN * BSZ * DMODEL) return;
    const int c = (int)(i % DMODEL);
    const int r = (int)(i / DMODEL);       // r = qi*2 + bi
    const int qi = r >> 1, bi = r & 1;
    const int tok = x[bi * QLEN + qi];
    E[i] = emb0[(long long)tok * DMODEL + c];
}

__global__ void k_posemb(float* __restrict__ pe) {
    long long i = (long long)blockIdx.x * blockDim.x + threadIdx.x;
    if (i >= (long long)KLEN * DMODEL) return;
    const int c = (int)(i % DMODEL);
    const int j = (int)(i / DMODEL);
    const float pos = fminf((float)(KLEN - 1 - j), 1000.0f);
    const int cc = (c < 384) ? c : c - 384;
    const float invf = __powf(10000.0f, -(2.0f * (float)cc) / (float)DMODEL);
    const float ang = pos * invf;
    pe[i] = (c < 384) ? __sinf(ang) : __cosf(ang);
}

// q prep: qac/qbd (bi,n,i,d) = q + r_w_bias / r_r_bias
__global__ void k_qprep(const float* __restrict__ heads, const float* __restrict__ rwb,
                        const float* __restrict__ rrb, float* __restrict__ qac,
                        float* __restrict__ qbd) {
    long long i = (long long)blockIdx.x * blockDim.x + threadIdx.x;
    if (i >= (long long)BSZ * NHEAD * QLEN * DHEAD) return;
    const int d  = (int)(i % DHEAD);
    const int qi = (int)((i / DHEAD) % QLEN);
    const int n  = (int)((i / ((long long)DHEAD * QLEN)) % NHEAD);
    const int bi = (int)(i / ((long long)DHEAD * QLEN * NHEAD));
    const float qv = heads[((long long)(MLEN + qi) * BSZ + bi) * (3 * DMODEL) + n * DHEAD + d];
    qac[i] = qv + rwb[n * DHEAD + d];
    qbd[i] = qv + rrb[n * DHEAD + d];
}

// k/v prep: K as (bi,n,j,d); V stored TRANSPOSED as (bi,n,d,j) so P@V is A*B^T too.
__global__ void k_kvprep(const float* __restrict__ heads, float* __restrict__ kb,
                         float* __restrict__ vbT) {
    long long i = (long long)blockIdx.x * blockDim.x + threadIdx.x;
    if (i >= (long long)BSZ * NHEAD * KLEN * DHEAD) return;
    const int d  = (int)(i % DHEAD);
    const int j  = (int)((i / DHEAD) % KLEN);
    const int n  = (int)((i / ((long long)DHEAD * KLEN)) % NHEAD);
    const int bi = (int)(i / ((long long)DHEAD * KLEN * NHEAD));
    const long long hrow = ((long long)j * BSZ + bi) * (3 * DMODEL);
    kb[i] = heads[hrow + DMODEL + n * DHEAD + d];
    vbT[(((long long)bi * NHEAD + n) * DHEAD + d) * KLEN + j] =
        heads[hrow + 2 * DMODEL + n * DHEAD + d];
}

// softmax over j for one (bi,n,i) row: score = (AC[j] + BDshift)*scale, causal mask,
// rel_shift folded in: out[i,j] = BD[i, j + (QLEN-1-i)]; OOB shift == masked positions.
__global__ __launch_bounds__(256)
void k_softmax(float* __restrict__ ac, const float* __restrict__ bd) {
    __shared__ float red[256];
    const int t = threadIdx.x;
    const int rowid = blockIdx.x;
    const int i  = rowid % QLEN;
    const int n  = (rowid / QLEN) % NHEAD;
    const int bi = rowid / (QLEN * NHEAD);
    const long long base = (((long long)bi * NHEAD + n) * QLEN + i) * KLEN;
    float* acr = ac + base;
    const float* bdr = bd + base;

    float vals[6];
    float mx = -3.0e38f;
    #pragma unroll
    for (int u = 0; u < 6; ++u) {
        const int j = t + u * 256;
        float s;
        if (j <= i + MLEN) {                 // causal: masked when j > i + mlen
            const int jd = j + (QLEN - 1 - i);
            const float b = (jd < KLEN) ? bdr[jd] : 0.0f;
            s = (acr[j] + b) * 0.125f;       // 1/sqrt(64)
        } else {
            s = -1.0e30f;
        }
        vals[u] = s;
        mx = fmaxf(mx, s);
    }
    red[t] = mx; __syncthreads();
    for (int off = 128; off > 0; off >>= 1) {
        if (t < off) red[t] = fmaxf(red[t], red[t + off]);
        __syncthreads();
    }
    mx = red[0]; __syncthreads();

    float sum = 0.0f;
    #pragma unroll
    for (int u = 0; u < 6; ++u) { vals[u] = __expf(vals[u] - mx); sum += vals[u]; }
    red[t] = sum; __syncthreads();
    for (int off = 128; off > 0; off >>= 1) {
        if (t < off) red[t] += red[t + off];
        __syncthreads();
    }
    const float inv = 1.0f / red[0];
    #pragma unroll
    for (int u = 0; u < 6; ++u) acr[t + u * 256] = vals[u] * inv;
}

// LayerNorm over 768; remap=1 writes dst row (bi*512+i) from src row (i*2+bi)
__global__ __launch_bounds__(256)
void k_layernorm(const float* __restrict__ x, const float* __restrict__ w,
                 const float* __restrict__ b, float* __restrict__ out, int remap) {
    __shared__ float red[256];
    const int t = threadIdx.x;
    const int r = blockIdx.x;
    const float x0 = x[(long long)r * DMODEL + t];
    const float x1 = x[(long long)r * DMODEL + t + 256];
    const float x2 = x[(long long)r * DMODEL + t + 512];
    red[t] = x0 + x1 + x2; __syncthreads();
    for (int off = 128; off > 0; off >>= 1) {
        if (t < off) red[t] += red[t + off];
        __syncthreads();
    }
    const float mean = red[0] * (1.0f / DMODEL); __syncthreads();
    const float d0 = x0 - mean, d1 = x1 - mean, d2 = x2 - mean;
    red[t] = d0 * d0 + d1 * d1 + d2 * d2; __syncthreads();
    for (int off = 128; off > 0; off >>= 1) {
        if (t < off) red[t] += red[t + off];
        __syncthreads();
    }
    const float rstd = rsqrtf(red[0] * (1.0f / DMODEL) + EPSF);
    const int dr = remap ? ((r & 1) * QLEN + (r >> 1)) : r;
    out[(long long)dr * DMODEL + t]       = d0 * rstd * w[t]       + b[t];
    out[(long long)dr * DMODEL + t + 256] = d1 * rstd * w[t + 256] + b[t + 256];
    out[(long long)dr * DMODEL + t + 512] = d2 * rstd * w[t + 512] + b[t + 512];
}

// ---------------- host-side helpers ----------------
static inline void gemm(const float* A, const float* B, const float* bias, const float* res,
                        float* C, int M, int N, int K, int lda, int ldb, int ldc,
                        long long sA1, long long sB1, long long sC1, int nb1,
                        long long sA2, long long sB2, long long sC2, int nb2,
                        int relu, float alpha, hipStream_t s) {
    dim3 g((N + BN - 1) / BN, (M + BM - 1) / BM, nb1 * nb2);
    k_gemm<<<g, 256, 0, s>>>(A, B, bias, res, C, M, N, K, lda, ldb, ldc,
                             sA1, sB1, sC1, nb1, sA2, sB2, sC2, relu, alpha);
}
static inline int nb256(long long n) { return (int)((n + 255) / 256); }

extern "C" void kernel_launch(void* const* d_in, const int* in_sizes, int n_in,
                              void* d_out, int out_size, void* d_ws, size_t ws_size,
                              hipStream_t stream) {
    (void)in_sizes; (void)n_in; (void)out_size; (void)ws_size;
    const int*   x      = (const int*)  d_in[0];
    const float* emb0   = (const float*)d_in[1];
    const float* proj0  = (const float*)d_in[2];
    // d_in[3..8]: emb1..proj3 — unused (all tokens < 249 fall in cluster 0)
    const float* qkv_w  = (const float*)d_in[9];
    const float* o_w    = (const float*)d_in[10];
    const float* r_w    = (const float*)d_in[11];
    const float* rwb    = (const float*)d_in[12];
    const float* rrb    = (const float*)d_in[13];
    const float* aln_w  = (const float*)d_in[14];
    const float* aln_b  = (const float*)d_in[15];
    const float* ff1w   = (const float*)d_in[16];
    const float* ff1b   = (const float*)d_in[17];
    const float* ff2w   = (const float*)d_in[18];
    const float* ff2b   = (const float*)d_in[19];
    const float* fln_w  = (const float*)d_in[20];
    const float* fln_b  = (const float*)d_in[21];
    const float* hln_w  = (const float*)d_in[22];
    const float* hln_b  = (const float*)d_in[23];
    const float* head_w = (const float*)d_in[24];
    const float* head_b = (const float*)d_in[25];

    float* ws  = (float*)d_ws;
    float* out = (float*)d_out;

    float* cat  = ws + O_CAT;
    float* hd   = ws + O_HEADS;
    float* qac  = ws + O_QAC;
    float* qbd  = ws + O_QBD;
    float* kb   = ws + O_KB;
    float* vbT  = ws + O_VB;
    float* rk   = ws + O_RK;
    float* acb  = ws + O_AC;
    float* bdb  = ws + O_BD;
    float* vec  = ws + O_VEC;
    float* h    = ws + O_H;
    float* tmp  = ws + O_TMP;
    float* ff   = ws + O_FF;
    float* E    = ws + O_FF;   // embed gather reuses FF scratch (disjoint in time)
    float* pe   = ws + O_PE;

    const long long HN = (long long)QLEN * BSZ * DMODEL;   // 786432
    const long long MEMS_BASE = (long long)BSZ * QLEN * WORDS; // 254976

    // --- embedding: h = gather(emb0) @ proj0^T * sqrt(d_model) ---
    k_gather_embed<<<nb256(HN), 256, 0, stream>>>(x, emb0, E);
    gemm(E, proj0, nullptr, nullptr, h, QLEN * BSZ, DMODEL, DMODEL,
         DMODEL, DMODEL, DMODEL, 0, 0, 0, 1, 0, 0, 0, 1, 0, 27.712812921102035f, stream);

    // --- sinusoidal relative positions (klen, d_model) ---
    k_posemb<<<nb256((long long)KLEN * DMODEL), 256, 0, stream>>>(pe);

    // --- cat = [zeros(mlen); h]; top half is always zeros (init mems) ---
    k_zero<<<nb256((long long)MLEN * BSZ * DMODEL), 256, 0, stream>>>(cat, (long long)MLEN * BSZ * DMODEL);

    for (int l = 0; l < NLAYER; ++l) {
        // new_mems[l] = [zeros(512,b,d); h]
        float* mems = out + MEMS_BASE + (long long)l * MLEN * BSZ * DMODEL;
        k_zero<<<nb256(HN), 256, 0, stream>>>(mems, HN);
        k_copy<<<nb256(HN), 256, 0, stream>>>(mems + HN, h, HN);
        // bottom of cat = current h
        k_copy<<<nb256(HN), 256, 0, stream>>>(cat + (long long)MLEN * BSZ * DMODEL, h, HN);

        // heads = cat @ qkv_w^T    (3072 x 2304)
        gemm(cat, qkv_w + (long long)l * 3 * DMODEL * DMODEL, nullptr, nullptr, hd,
             KLEN * BSZ, 3 * DMODEL, DMODEL, DMODEL, DMODEL, 3 * DMODEL,
             0, 0, 0, 1, 0, 0, 0, 1, 0, 1.0f, stream);

        // q (+biases), k, v rearranged per head
        k_qprep<<<nb256((long long)BSZ * NHEAD * QLEN * DHEAD), 256, 0, stream>>>(
            hd, rwb + (long long)l * DMODEL, rrb + (long long)l * DMODEL, qac, qbd);
        k_kvprep<<<nb256((long long)BSZ * NHEAD * KLEN * DHEAD), 256, 0, stream>>>(hd, kb, vbT);

        // r_k = pos_emb @ r_w^T  (1536 x 768)
        gemm(pe, r_w + (long long)l * DMODEL * DMODEL, nullptr, nullptr, rk,
             KLEN, DMODEL, DMODEL, DMODEL, DMODEL, DMODEL,
             0, 0, 0, 1, 0, 0, 0, 1, 0, 1.0f, stream);

        // AC = (q + r_w_bias) @ k^T  batched over (n inner=12, bi outer=2)
        gemm(qac, kb, nullptr, nullptr, acb, QLEN, KLEN, DHEAD,
             DHEAD, DHEAD, KLEN,
             (long long)QLEN * DHEAD, (long long)KLEN * DHEAD, (long long)QLEN * KLEN, NHEAD,
             (long long)NHEAD * QLEN * DHEAD, (long long)NHEAD * KLEN * DHEAD,
             (long long)NHEAD * QLEN * KLEN, BSZ, 0, 1.0f, stream);

        // BD = (q + r_r_bias) @ r_k^T  (r_k shared across batch bi)
        gemm(qbd, rk, nullptr, nullptr, bdb, QLEN, KLEN, DHEAD,
             DHEAD, DMODEL, KLEN,
             (long long)QLEN * DHEAD, (long long)DHEAD, (long long)QLEN * KLEN, NHEAD,
             (long long)NHEAD * QLEN * DHEAD, 0LL,
             (long long)NHEAD * QLEN * KLEN, BSZ, 0, 1.0f, stream);

        // softmax((AC + shift(BD)) * scale) with causal mask; prob overwrites AC
        k_softmax<<<QLEN * NHEAD * BSZ, 256, 0, stream>>>(acb, bdb);

        // vec = prob @ V: with V stored (d, j), this is A(512x1536) * B(64,1536)^T
        gemm(acb, vbT, nullptr, nullptr, vec, QLEN, DHEAD, KLEN,
             KLEN, KLEN, BSZ * DMODEL,
             (long long)QLEN * KLEN, (long long)DHEAD * KLEN, (long long)DHEAD, NHEAD,
             (long long)NHEAD * QLEN * KLEN, (long long)NHEAD * DHEAD * KLEN,
             (long long)DMODEL, BSZ, 0, 1.0f, stream);

        // attn out + residual, then LN
        gemm(vec, o_w + (long long)l * DMODEL * DMODEL, nullptr, h, tmp,
             QLEN * BSZ, DMODEL, DMODEL, DMODEL, DMODEL, DMODEL,
             0, 0, 0, 1, 0, 0, 0, 1, 0, 1.0f, stream);
        k_layernorm<<<QLEN * BSZ, 256, 0, stream>>>(tmp, aln_w + (long long)l * DMODEL,
                                                    aln_b + (long long)l * DMODEL, h, 0);

        // FFN: relu(h @ ff1^T + b1) @ ff2^T + b2 + h, then LN
        gemm(h, ff1w + (long long)l * DINNER * DMODEL, ff1b + (long long)l * DINNER, nullptr, ff,
             QLEN * BSZ, DINNER, DMODEL, DMODEL, DMODEL, DINNER,
             0, 0, 0, 1, 0, 0, 0, 1, 1, 1.0f, stream);
        gemm(ff, ff2w + (long long)l * DMODEL * DINNER, ff2b + (long long)l * DMODEL, h, tmp,
             QLEN * BSZ, DMODEL, DINNER, DINNER, DINNER, DMODEL,
             0, 0, 0, 1, 0, 0, 0, 1, 0, 1.0f, stream);
        k_layernorm<<<QLEN * BSZ, 256, 0, stream>>>(tmp, fln_w + (long long)l * DMODEL,
                                                    fln_b + (long long)l * DMODEL, h, 0);
    }

    // final: LN with transpose to (bi, i, d), then logits = lh @ head_w^T + head_b
    k_layernorm<<<QLEN * BSZ, 256, 0, stream>>>(h, hln_w, hln_b, tmp, 1);
    gemm(tmp, head_w, head_b, nullptr, out, QLEN * BSZ, WORDS, DMODEL,
         DMODEL, DMODEL, WORDS, 0, 0, 0, 1, 0, 0, 0, 1, 0, 1.0f, stream);
}
